// MultiHeadAttention_39573828665764
// MI455X (gfx1250) — matrix operand order repaired and checked by reading, not compiled, not run on hardware
//
#include <hip/hip_runtime.h>

#define BQ     4
#define NSEQ   1024
#define DMODEL 1024
#define NHEAD  16
#define DHEAD  64
#define NBH    (BQ * NHEAD)   // 64
#define MROWS  (BQ * NSEQ)    // 4096
#define EPSF   1e-8f
#define LDP    40             // padded LDS row stride for 32-wide k tiles (elems)

typedef __attribute__((ext_vector_type(16))) __bf16    bf16x16;
typedef __attribute__((ext_vector_type(8)))  float     f32x8;
typedef __attribute__((ext_vector_type(4)))  unsigned  u32x4;

__device__ __forceinline__ unsigned f2u(float f)   { union { float f; unsigned u; } v{f}; return v.u; }
__device__ __forceinline__ float    u2f(unsigned u){ union { unsigned u; float f; } v{u}; return v.f; }
__device__ __forceinline__ __bf16   bits2bf(unsigned short s) { union { unsigned short u; __bf16 b; } v{s}; return v.b; }
__device__ __forceinline__ float    bf2f(__bf16 b) { union { __bf16 b; unsigned short u; } v{b}; return u2f(((unsigned)v.u) << 16); }

// Split fp32 into bf16 hi + bf16 lo (RNE); hi+lo ~= x to ~2^-18 relative.
__device__ __forceinline__ void split_bf16(float x, __bf16& hi, __bf16& lo) {
  unsigned u = f2u(x);
  unsigned short h = (unsigned short)((u + 0x7FFFu + ((u >> 16) & 1u)) >> 16);
  float xh = u2f(((unsigned)h) << 16);
  unsigned ul = f2u(x - xh);
  unsigned short l = (unsigned short)((ul + 0x7FFFu + ((ul >> 16) & 1u)) >> 16);
  hi = bits2bf(h);
  lo = bits2bf(l);
}

union FragU { bf16x16 v; u32x4 q[2]; };

// ---- LDS staging (256 threads), tiles are [rows][32] padded to stride LDP ---
// 128x32 bf16 tile from global row-major (ldg elems)
__device__ __forceinline__ void stage128(const __bf16* __restrict__ g, int ldg,
                                         __bf16* s) {
  int t = threadIdx.x;
#pragma unroll
  for (int i = 0; i < 2; ++i) {
    int c = t + 256 * i;          // 512 chunks of 8 elems
    int row = c >> 2, col = (c & 3) * 8;
    *(u32x4*)(s + row * LDP + col) = *(const u32x4*)(g + (size_t)row * ldg + col);
  }
}
// 64x32 bf16 tile
__device__ __forceinline__ void stage64(const __bf16* __restrict__ g, int ldg,
                                        __bf16* s) {
  int c = threadIdx.x;            // 256 chunks of 8 elems
  int row = c >> 2, col = (c & 3) * 8;
  *(u32x4*)(s + row * LDP + col) = *(const u32x4*)(g + (size_t)row * ldg + col);
}
// 128x32 fp32 tile -> split into hi/lo LDS tiles (one-shot conversion of P)
__device__ __forceinline__ void stage128_f32(const float* __restrict__ g, int ldg,
                                             __bf16* shi, __bf16* slo) {
  int t = threadIdx.x;
#pragma unroll
  for (int i = 0; i < 2; ++i) {
    int c = t + 256 * i;
    int row = c >> 2, col = (c & 3) * 8;
    const float* gp = g + (size_t)row * ldg + col;
#pragma unroll
    for (int j = 0; j < 8; ++j) {
      __bf16 h, l; split_bf16(gp[j], h, l);
      shi[row * LDP + col + j] = h;
      slo[row * LDP + col + j] = l;
    }
  }
}

// ---- LDS fragment loads (ds_load_b128) --------------------------------------
// A fragment 16x32: lane M=lane&15; elem j: K = j + 8*((j>>3)+(lane>>4))
__device__ __forceinline__ bf16x16 lds_a_frag(const __bf16* p0) {
  int lane = threadIdx.x & 31;
  const __bf16* p = p0 + (lane & 15) * LDP + (lane >> 4) * 8;
  FragU u;
  u.q[0] = *(const u32x4*)(p);
  u.q[1] = *(const u32x4*)(p + 16);
  return u.v;
}
// B fragment 32x16 from [n][k] tile: lane N=lane&15; elem j: K = j + 8*((j>>3)+(lane>>4))
__device__ __forceinline__ bf16x16 lds_b_frag(const __bf16* p0) {
  int lane = threadIdx.x & 31;
  const __bf16* p = p0 + (lane & 15) * LDP + (lane >> 4) * 8;
  FragU u;
  u.q[0] = *(const u32x4*)(p);
  u.q[1] = *(const u32x4*)(p + 16);
  return u.v;
}

// Compensated product: acc += Ah*Bh + Ah*Bl + Al*Bh
__device__ __forceinline__ void mma3(f32x8& acc, const bf16x16& ah, const bf16x16& al,
                                     const bf16x16& bh, const bf16x16& bl) {
  acc = __builtin_amdgcn_wmma_f32_16x16x32_bf16(false, ah, false, bh, (short)0, acc, false, false);
  acc = __builtin_amdgcn_wmma_f32_16x16x32_bf16(false, ah, false, bl, (short)0, acc, false, false);
  acc = __builtin_amdgcn_wmma_f32_16x16x32_bf16(false, al, false, bh, (short)0, acc, false, false);
}

// ---------------------------------------------------------------------------
// 0a) Elementwise fp32 -> bf16 hi/lo planes
__global__ void __launch_bounds__(256) cvt_plane_kernel(
    const float* __restrict__ src, __bf16* __restrict__ hi, __bf16* __restrict__ lo, int n) {
  int i = blockIdx.x * 256 + threadIdx.x;
  if (i < n) { __bf16 h, l; split_bf16(src[i], h, l); hi[i] = h; lo[i] = l; }
}
// 0b) 1024x1024 fp32 -> transposed bf16 hi/lo planes (W[k][n] -> T[n][k])
__global__ void __launch_bounds__(256) cvt_transpose_kernel(
    const float* __restrict__ W, __bf16* __restrict__ thi, __bf16* __restrict__ tlo) {
  int i = blockIdx.x * 256 + threadIdx.x;
  int r = i >> 10, c = i & 1023;
  __bf16 h, l; split_bf16(W[i], h, l);
  thi[(size_t)c * DMODEL + r] = h;
  tlo[(size_t)c * DMODEL + r] = l;
}

// ---------------------------------------------------------------------------
// 1) Projections: X[4096,1024] @ W[1024,1024]; z: 0=Q(scale 1/8), 1=K, 2=V.
//    Block 256 thr, tile 128x128; wave tile 64x32 (8 waves 2x4). LDS-staged.
__global__ void __launch_bounds__(256) proj_kernel(
    const __bf16* __restrict__ xhi, const __bf16* __restrict__ xlo,
    const __bf16* __restrict__ wthi, const __bf16* __restrict__ wtlo,
    __bf16* __restrict__ qhhi, __bf16* __restrict__ qhlo,
    __bf16* __restrict__ khhi, __bf16* __restrict__ khlo,
    __bf16* __restrict__ vthi, __bf16* __restrict__ vtlo) {
  __shared__ __attribute__((aligned(16))) __bf16 As_hi[128 * LDP];
  __shared__ __attribute__((aligned(16))) __bf16 As_lo[128 * LDP];
  __shared__ __attribute__((aligned(16))) __bf16 Bs_hi[128 * LDP];
  __shared__ __attribute__((aligned(16))) __bf16 Bs_lo[128 * LDP];

  int z = blockIdx.z;
  const __bf16* Ahi = xhi + (size_t)z * MROWS * DMODEL;
  const __bf16* Alo = xlo + (size_t)z * MROWS * DMODEL;
  const __bf16* Bhi = wthi + (size_t)z * DMODEL * DMODEL;
  const __bf16* Blo = wtlo + (size_t)z * DMODEL * DMODEL;
  float scale = (z == 0) ? 0.125f : 1.0f;

  int row0 = blockIdx.y * 128;
  int col0 = blockIdx.x * 128;
  int wave = threadIdx.x >> 5;
  int wrow = (wave >> 2) * 64;   // 0 / 64
  int wcol = (wave & 3) * 32;    // 0,32,64,96

  f32x8 acc[4][2] = {};
  for (int k0 = 0; k0 < DMODEL; k0 += 32) {
    if (k0) __syncthreads();
    stage128(Ahi + (size_t)row0 * DMODEL + k0, DMODEL, As_hi);
    stage128(Alo + (size_t)row0 * DMODEL + k0, DMODEL, As_lo);
    stage128(Bhi + (size_t)col0 * DMODEL + k0, DMODEL, Bs_hi);
    stage128(Blo + (size_t)col0 * DMODEL + k0, DMODEL, Bs_lo);
    __syncthreads();
    bf16x16 ah[4], al[4], bh[2], bl[2];
#pragma unroll
    for (int ti = 0; ti < 4; ++ti) {
      ah[ti] = lds_a_frag(As_hi + (wrow + ti * 16) * LDP);
      al[ti] = lds_a_frag(As_lo + (wrow + ti * 16) * LDP);
    }
#pragma unroll
    for (int tj = 0; tj < 2; ++tj) {
      bh[tj] = lds_b_frag(Bs_hi + (wcol + tj * 16) * LDP);
      bl[tj] = lds_b_frag(Bs_lo + (wcol + tj * 16) * LDP);
    }
#pragma unroll
    for (int ti = 0; ti < 4; ++ti)
#pragma unroll
      for (int tj = 0; tj < 2; ++tj)
        mma3(acc[ti][tj], ah[ti], al[ti], bh[tj], bl[tj]);
  }

  int lane = threadIdx.x & 31;
  int cn = lane & 15, sel = lane >> 4;
#pragma unroll
  for (int ti = 0; ti < 4; ++ti)
#pragma unroll
    for (int tj = 0; tj < 2; ++tj)
#pragma unroll
      for (int vv = 0; vv < 8; ++vv) {
        int m = row0 + wrow + ti * 16 + vv + sel * 8;
        int n = col0 + wcol + tj * 16 + cn;
        int b = m >> 10, qi = m & 1023;
        int hd = n >> 6, d = n & 63;
        int bh2 = b * NHEAD + hd;
        __bf16 h, l; split_bf16(acc[ti][tj][vv] * scale, h, l);
        if (z == 0) {
          size_t idx = ((size_t)bh2 * NSEQ + qi) * DHEAD + d;
          qhhi[idx] = h; qhlo[idx] = l;
        } else if (z == 1) {
          size_t idx = ((size_t)bh2 * NSEQ + qi) * DHEAD + d;
          khhi[idx] = h; khlo[idx] = l;
        } else {
          size_t idx = ((size_t)bh2 * DHEAD + d) * NSEQ + qi;  // transposed
          vthi[idx] = h; vtlo[idx] = l;
        }
      }
}

// ---------------------------------------------------------------------------
// 2) Logits: per (b,h): S[q,k] = (Qh . Kh^T) * weight, masked. Tile 128x128.
__global__ void __launch_bounds__(256) logits_kernel(
    const __bf16* __restrict__ qhhi, const __bf16* __restrict__ qhlo,
    const __bf16* __restrict__ khhi, const __bf16* __restrict__ khlo,
    const float* __restrict__ weight, const unsigned char* __restrict__ mask,
    float* __restrict__ S) {
  __shared__ __attribute__((aligned(16))) __bf16 As_hi[128 * LDP];
  __shared__ __attribute__((aligned(16))) __bf16 As_lo[128 * LDP];
  __shared__ __attribute__((aligned(16))) __bf16 Bs_hi[128 * LDP];
  __shared__ __attribute__((aligned(16))) __bf16 Bs_lo[128 * LDP];

  int bh = blockIdx.z;
  int b = bh >> 4;
  size_t hofs = (size_t)bh * NSEQ * DHEAD;
  int row0 = blockIdx.y * 128;   // q
  int col0 = blockIdx.x * 128;   // k
  int wave = threadIdx.x >> 5;
  int wrow = (wave >> 2) * 64;
  int wcol = (wave & 3) * 32;

  f32x8 acc[4][2] = {};
#pragma unroll
  for (int k0 = 0; k0 < DHEAD; k0 += 32) {
    if (k0) __syncthreads();
    stage128(qhhi + hofs + (size_t)row0 * DHEAD + k0, DHEAD, As_hi);
    stage128(qhlo + hofs + (size_t)row0 * DHEAD + k0, DHEAD, As_lo);
    stage128(khhi + hofs + (size_t)col0 * DHEAD + k0, DHEAD, Bs_hi);
    stage128(khlo + hofs + (size_t)col0 * DHEAD + k0, DHEAD, Bs_lo);
    __syncthreads();
    bf16x16 ah[4], al[4], bhv[2], blv[2];
#pragma unroll
    for (int ti = 0; ti < 4; ++ti) {
      ah[ti] = lds_a_frag(As_hi + (wrow + ti * 16) * LDP);
      al[ti] = lds_a_frag(As_lo + (wrow + ti * 16) * LDP);
    }
#pragma unroll
    for (int tj = 0; tj < 2; ++tj) {
      bhv[tj] = lds_b_frag(Bs_hi + (wcol + tj * 16) * LDP);
      blv[tj] = lds_b_frag(Bs_lo + (wcol + tj * 16) * LDP);
    }
#pragma unroll
    for (int ti = 0; ti < 4; ++ti)
#pragma unroll
      for (int tj = 0; tj < 2; ++tj)
        mma3(acc[ti][tj], ah[ti], al[ti], bhv[tj], blv[tj]);
  }

  int lane = threadIdx.x & 31;
  int cn = lane & 15, sel = lane >> 4;
#pragma unroll
  for (int ti = 0; ti < 4; ++ti)
#pragma unroll
    for (int tj = 0; tj < 2; ++tj)
#pragma unroll
      for (int vv = 0; vv < 8; ++vv) {
        int qi = row0 + wrow + ti * 16 + vv + sel * 8;
        int ki = col0 + wcol + tj * 16 + cn;
        size_t widx = ((size_t)b * NSEQ + qi) * NSEQ + ki;
        float val = acc[ti][tj][vv] * weight[widx];
        if (mask[widx]) val = -1e30f;
        S[((size_t)bh * NSEQ + qi) * NSEQ + ki] = val;
      }
}

// ---------------------------------------------------------------------------
// 3) Column stats for softmax over the QUERY axis.
__global__ void __launch_bounds__(256) colstats_kernel(
    const float* __restrict__ S, float* __restrict__ colmax, float* __restrict__ colsum) {
  int bh = blockIdx.y;
  int kk = blockIdx.x * 256 + threadIdx.x;
  const float* Sp = S + (size_t)bh * NSEQ * NSEQ + kk;
  float m = -1e38f, s = 0.f;
  for (int qi = 0; qi < NSEQ; ++qi) {
    float x = Sp[(size_t)qi * NSEQ];
    if (x > m) { s *= __expf(m - x); m = x; }
    s += __expf(x - m);
  }
  colmax[bh * NSEQ + kk] = m;
  colsum[bh * NSEQ + kk] = s;
}

// 3b) Vsum[bh][d] = sum_k Vh[bh][k][d] from transposed planes (hi+lo).
__global__ void __launch_bounds__(64) vsum_kernel(
    const __bf16* __restrict__ vthi, const __bf16* __restrict__ vtlo,
    float* __restrict__ Vsum) {
  int bh = blockIdx.x, d = threadIdx.x;
  const __bf16* ph = vthi + ((size_t)bh * DHEAD + d) * NSEQ;
  const __bf16* pl = vtlo + ((size_t)bh * DHEAD + d) * NSEQ;
  float s = 0.f;
  for (int kk = 0; kk < NSEQ; ++kk) s += bf2f(ph[kk]) + bf2f(pl[kk]);
  Vsum[bh * DHEAD + d] = s;
}

// ---------------------------------------------------------------------------
// 4) In-place p = exp(S - colmax)/colsum, and inv_rowsum = 1/(sum_k p + N*EPS)
__global__ void __launch_bounds__(256) prob_kernel(
    float* __restrict__ S, const float* __restrict__ colmax,
    const float* __restrict__ colsum, float* __restrict__ inv_rowsum) {
  __shared__ float red[256];
  int bh = blockIdx.y, qi = blockIdx.x;
  float* row = S + ((size_t)bh * NSEQ + qi) * NSEQ;
  const float* cm = colmax + bh * NSEQ;
  const float* cs = colsum + bh * NSEQ;
  float part = 0.f;
  for (int kk = threadIdx.x; kk < NSEQ; kk += 256) {
    float p = __expf(row[kk] - cm[kk]) / cs[kk];
    row[kk] = p;
    part += p;
  }
  red[threadIdx.x] = part;
  __syncthreads();
  for (int off = 128; off > 0; off >>= 1) {
    if (threadIdx.x < off) red[threadIdx.x] += red[threadIdx.x + off];
    __syncthreads();
  }
  if (threadIdx.x == 0)
    inv_rowsum[bh * NSEQ + qi] = 1.f / (red[0] + (float)NSEQ * EPSF);
}

// ---------------------------------------------------------------------------
// 5) out = P @ Vh (M=1024,N=64,K=1024). Block tile 128x64, wave tile 32x32.
//    P is converted fp32->bf16 hi/lo during staging (once per element).
__global__ void __launch_bounds__(256) av_kernel(
    const float* __restrict__ S,
    const __bf16* __restrict__ vthi, const __bf16* __restrict__ vtlo,
    const float* __restrict__ Vsum, const float* __restrict__ inv_rowsum,
    __bf16* __restrict__ ohi, __bf16* __restrict__ olo) {
  __shared__ __attribute__((aligned(16))) __bf16 As_hi[128 * LDP];
  __shared__ __attribute__((aligned(16))) __bf16 As_lo[128 * LDP];
  __shared__ __attribute__((aligned(16))) __bf16 Bs_hi[64 * LDP];
  __shared__ __attribute__((aligned(16))) __bf16 Bs_lo[64 * LDP];

  int bh = blockIdx.z;
  int b = bh >> 4, hd = bh & 15;
  const float* P = S + (size_t)bh * NSEQ * NSEQ;
  size_t vofs = (size_t)bh * DHEAD * NSEQ;
  int row0 = blockIdx.y * 128;
  int wave = threadIdx.x >> 5;
  int wrow = (wave >> 1) * 32;   // 0,32,64,96
  int wcol = (wave & 1) * 32;    // 0,32

  f32x8 acc[2][2] = {};
  for (int k0 = 0; k0 < NSEQ; k0 += 32) {
    if (k0) __syncthreads();
    stage128_f32(P + (size_t)row0 * NSEQ + k0, NSEQ, As_hi, As_lo);
    stage64(vthi + vofs + k0, NSEQ, Bs_hi);
    stage64(vtlo + vofs + k0, NSEQ, Bs_lo);
    __syncthreads();
    bf16x16 ah[2], al[2], bhv[2], blv[2];
#pragma unroll
    for (int ti = 0; ti < 2; ++ti) {
      ah[ti] = lds_a_frag(As_hi + (wrow + ti * 16) * LDP);
      al[ti] = lds_a_frag(As_lo + (wrow + ti * 16) * LDP);
    }
#pragma unroll
    for (int tj = 0; tj < 2; ++tj) {
      bhv[tj] = lds_b_frag(Bs_hi + (wcol + tj * 16) * LDP);
      blv[tj] = lds_b_frag(Bs_lo + (wcol + tj * 16) * LDP);
    }
#pragma unroll
    for (int ti = 0; ti < 2; ++ti)
#pragma unroll
      for (int tj = 0; tj < 2; ++tj)
        mma3(acc[ti][tj], ah[ti], al[ti], bhv[tj], blv[tj]);
  }

  int lane = threadIdx.x & 31;
  int cn = lane & 15, sel = lane >> 4;
#pragma unroll
  for (int ti = 0; ti < 2; ++ti)
#pragma unroll
    for (int tj = 0; tj < 2; ++tj)
#pragma unroll
      for (int vv = 0; vv < 8; ++vv) {
        int qi = row0 + wrow + ti * 16 + vv + sel * 8;
        int d = wcol + tj * 16 + cn;
        float val = (acc[ti][tj][vv] + EPSF * Vsum[bh * DHEAD + d]) * inv_rowsum[bh * NSEQ + qi];
        size_t idx = ((size_t)b * NSEQ + qi) * DMODEL + hd * DHEAD + d;
        __bf16 h, l; split_bf16(val, h, l);
        ohi[idx] = h; olo[idx] = l;
      }
}

// ---------------------------------------------------------------------------
// 6) Final: Y = O[4096,1024] @ Wo[1024,1024] -> d_out (fp32). LDS-staged.
__global__ void __launch_bounds__(256) outproj_kernel(
    const __bf16* __restrict__ ohi, const __bf16* __restrict__ olo,
    const __bf16* __restrict__ wothi, const __bf16* __restrict__ wotlo,
    float* __restrict__ Y) {
  __shared__ __attribute__((aligned(16))) __bf16 As_hi[128 * LDP];
  __shared__ __attribute__((aligned(16))) __bf16 As_lo[128 * LDP];
  __shared__ __attribute__((aligned(16))) __bf16 Bs_hi[128 * LDP];
  __shared__ __attribute__((aligned(16))) __bf16 Bs_lo[128 * LDP];

  int row0 = blockIdx.y * 128;
  int col0 = blockIdx.x * 128;
  int wave = threadIdx.x >> 5;
  int wrow = (wave >> 2) * 64;
  int wcol = (wave & 3) * 32;

  f32x8 acc[4][2] = {};
  for (int k0 = 0; k0 < DMODEL; k0 += 32) {
    if (k0) __syncthreads();
    stage128(ohi + (size_t)row0 * DMODEL + k0, DMODEL, As_hi);
    stage128(olo + (size_t)row0 * DMODEL + k0, DMODEL, As_lo);
    stage128(wothi + (size_t)col0 * DMODEL + k0, DMODEL, Bs_hi);
    stage128(wotlo + (size_t)col0 * DMODEL + k0, DMODEL, Bs_lo);
    __syncthreads();
    bf16x16 ah[4], al[4], bhv[2], blv[2];
#pragma unroll
    for (int ti = 0; ti < 4; ++ti) {
      ah[ti] = lds_a_frag(As_hi + (wrow + ti * 16) * LDP);
      al[ti] = lds_a_frag(As_lo + (wrow + ti * 16) * LDP);
    }
#pragma unroll
    for (int tj = 0; tj < 2; ++tj) {
      bhv[tj] = lds_b_frag(Bs_hi + (wcol + tj * 16) * LDP);
      blv[tj] = lds_b_frag(Bs_lo + (wcol + tj * 16) * LDP);
    }
#pragma unroll
    for (int ti = 0; ti < 4; ++ti)
#pragma unroll
      for (int tj = 0; tj < 2; ++tj)
        mma3(acc[ti][tj], ah[ti], al[ti], bhv[tj], blv[tj]);
  }

  int lane = threadIdx.x & 31;
  int cn = lane & 15, sel = lane >> 4;
#pragma unroll
  for (int ti = 0; ti < 4; ++ti)
#pragma unroll
    for (int tj = 0; tj < 2; ++tj)
#pragma unroll
      for (int vv = 0; vv < 8; ++vv) {
        int m = row0 + wrow + ti * 16 + vv + sel * 8;
        int n = col0 + wcol + tj * 16 + cn;
        Y[(size_t)m * DMODEL + n] = acc[ti][tj][vv];
      }
}

extern "C" void kernel_launch(void* const* d_in, const int* in_sizes, int n_in,
                              void* d_out, int out_size, void* d_ws, size_t ws_size,
                              hipStream_t stream) {
  const float* q  = (const float*)d_in[0];
  const float* k  = (const float*)d_in[1];
  const float* v  = (const float*)d_in[2];
  const unsigned char* mask = (const unsigned char*)d_in[3];
  const float* weight = (const float*)d_in[4];
  const float* Wq = (const float*)d_in[5];
  const float* Wk = (const float*)d_in[6];
  const float* Wv = (const float*)d_in[7];
  const float* Wo = (const float*)d_in[8];
  float* Y = (float*)d_out;

  const size_t NX  = (size_t)MROWS * DMODEL;     // 4M elems
  const size_t NW  = (size_t)DMODEL * DMODEL;    // 1M elems
  const size_t NHS = (size_t)NBH * NSEQ * DHEAD; // 4M elems

  char* cur = (char*)d_ws;
  float* S          = (float*)cur; cur += (size_t)NBH * NSEQ * NSEQ * 4;   // 256MB
  float* colmax     = (float*)cur; cur += (size_t)NBH * NSEQ * 4;
  float* colsum     = (float*)cur; cur += (size_t)NBH * NSEQ * 4;
  float* inv_rowsum = (float*)cur; cur += (size_t)NBH * NSEQ * 4;
  float* Vsum       = (float*)cur; cur += (size_t)NBH * DHEAD * 4;
  __bf16* xhi   = (__bf16*)cur; cur += 3 * NX * 2;
  __bf16* xlo   = (__bf16*)cur; cur += 3 * NX * 2;
  __bf16* wthi  = (__bf16*)cur; cur += 3 * NW * 2;
  __bf16* wtlo  = (__bf16*)cur; cur += 3 * NW * 2;
  __bf16* wothi = (__bf16*)cur; cur += NW * 2;
  __bf16* wotlo = (__bf16*)cur; cur += NW * 2;
  __bf16* qhhi  = (__bf16*)cur; cur += NHS * 2;
  __bf16* qhlo  = (__bf16*)cur; cur += NHS * 2;
  __bf16* khhi  = (__bf16*)cur; cur += NHS * 2;
  __bf16* khlo  = (__bf16*)cur; cur += NHS * 2;
  __bf16* vthi  = (__bf16*)cur; cur += NHS * 2;
  __bf16* vtlo  = (__bf16*)cur; cur += NHS * 2;
  __bf16* ohi   = (__bf16*)cur; cur += NX * 2;
  __bf16* olo   = (__bf16*)cur; cur += NX * 2;

  int nblk = (int)(NX / 256);
  cvt_plane_kernel<<<nblk, 256, 0, stream>>>(q, xhi + 0 * NX, xlo + 0 * NX, (int)NX);
  cvt_plane_kernel<<<nblk, 256, 0, stream>>>(k, xhi + 1 * NX, xlo + 1 * NX, (int)NX);
  cvt_plane_kernel<<<nblk, 256, 0, stream>>>(v, xhi + 2 * NX, xlo + 2 * NX, (int)NX);
  int wblk = (int)(NW / 256);
  cvt_transpose_kernel<<<wblk, 256, 0, stream>>>(Wq, wthi + 0 * NW, wtlo + 0 * NW);
  cvt_transpose_kernel<<<wblk, 256, 0, stream>>>(Wk, wthi + 1 * NW, wtlo + 1 * NW);
  cvt_transpose_kernel<<<wblk, 256, 0, stream>>>(Wv, wthi + 2 * NW, wtlo + 2 * NW);
  cvt_transpose_kernel<<<wblk, 256, 0, stream>>>(Wo, wothi, wotlo);

  dim3 blk(256);
  proj_kernel<<<dim3(DMODEL / 128, MROWS / 128, 3), blk, 0, stream>>>(
      xhi, xlo, wthi, wtlo, qhhi, qhlo, khhi, khlo, vthi, vtlo);
  logits_kernel<<<dim3(NSEQ / 128, NSEQ / 128, NBH), blk, 0, stream>>>(
      qhhi, qhlo, khhi, khlo, weight, mask, S);
  colstats_kernel<<<dim3(NSEQ / 256, NBH), 256, 0, stream>>>(S, colmax, colsum);
  vsum_kernel<<<dim3(NBH), 64, 0, stream>>>(vthi, vtlo, Vsum);
  prob_kernel<<<dim3(NSEQ, NBH), 256, 0, stream>>>(S, colmax, colsum, inv_rowsum);
  av_kernel<<<dim3(1, NSEQ / 128, NBH), blk, 0, stream>>>(
      S, vthi, vtlo, Vsum, inv_rowsum, ohi, olo);
  outproj_kernel<<<dim3(DMODEL / 128, MROWS / 128, 1), blk, 0, stream>>>(
      ohi, olo, wothi, wotlo, Y);
}
